// SpiralConv_9878424780834
// MI455X (gfx1250) — matrix operand order repaired and checked by reading, not compiled, not run on hardware
//
#include <hip/hip_runtime.h>
#include <stdint.h>

// ---------------------------------------------------------------------------
// SpiralConv: y[b,n,o] = ELU( sum_{s,f} x[b, adj[b,n,s], f] * W[o, s*F+f] + b[o] )
// B=8, N=50000, F=32, S=16, OUT=64 -> gather + GEMM (M=400k, N=64, K=512)
// CDNA5 strategy:
//   * pre-convert x and W to bf16 in d_ws
//   * TDM gather-mode (tensor_load_to_lds, 16 x 16-bit row indices) assembles
//     each 16-neighbor A-tile row (1 KB) directly into LDS  [TENSORcnt]
//   * v_wmma_f32_16x16x32_bf16, f32 accumulate; B fragments are 32 contiguous
//     bytes of row-major W (L0/L2 resident)
// ---------------------------------------------------------------------------

typedef __attribute__((ext_vector_type(16))) __bf16 v16bf;
typedef __attribute__((ext_vector_type(8)))  __bf16 v8bf;
typedef __attribute__((ext_vector_type(8)))  float  v8f;
typedef __attribute__((ext_vector_type(4)))  unsigned int u32x4;
typedef __attribute__((ext_vector_type(8)))  int i32x8;
typedef __attribute__((ext_vector_type(4)))  int i32x4;

#define B_    8
#define N_    50000
#define F_    32
#define S_    16
#define OUT_  64
#define K_    (F_ * S_)            // 512
#define TILES_PER_BATCH (N_ / 16)  // 3125 (exact)
#define LDS_STRIDE 520             // elements; 1040 B row stride (16B aligned, bank-rotated)

// float -> bf16, round-to-nearest-even
__device__ __forceinline__ unsigned short f2bf(float f) {
    union { float f; unsigned u; } v; v.f = f;
    unsigned u = v.u;
    unsigned r = (u + 0x7FFFu + ((u >> 16) & 1u)) >> 16;
    return (unsigned short)r;
}

// --- pre-pass: W (64x512 f32) -> bf16 (64 KB) --------------------------------
__global__ void convert_w_kernel(const float* __restrict__ W,
                                 unsigned short* __restrict__ Wbf) {
    int i = blockIdx.x * 256 + threadIdx.x;   // 32768 total, exact grid
    Wbf[i] = f2bf(W[i]);
}

// --- pre-pass: x (B*N*F f32) -> bf16 -----------------------------------------
__global__ void convert_x_kernel(const float* __restrict__ x,
                                 unsigned short* __restrict__ xbf, int n4) {
    int i = blockIdx.x * 256 + threadIdx.x;
    if (i < n4) {
        float4 v = ((const float4*)x)[i];
        ushort4 o;
        o.x = f2bf(v.x); o.y = f2bf(v.y); o.z = f2bf(v.z); o.w = f2bf(v.w);
        ((ushort4*)xbf)[i] = o;
    }
}

// --- TDM gather: one descriptor pulls 16 rows (16-bit indices) of the
//     (N_ x F_) bf16 tensor into 1024 contiguous LDS bytes at lds_addr --------
__device__ __forceinline__ void tdm_gather_row(unsigned lds_addr,
                                               unsigned long long gaddr,
                                               const int* __restrict__ arow) {
    int4 qa = ((const int4*)arow)[0];
    int4 qb = ((const int4*)arow)[1];
    int4 qc = ((const int4*)arow)[2];
    int4 qd = ((const int4*)arow)[3];

    u32x4 g0;
    g0.x = 0x80000001u;                               // count=1, gather_mode=1, idx16
    g0.y = lds_addr;                                  // LDS byte address of tile
    g0.z = (unsigned)(gaddr & 0xFFFFFFFFu);           // global_addr[31:0]
    g0.w = (unsigned)((gaddr >> 32) & 0x01FFFFFFu)    // global_addr[56:32]
           | (2u << 30);                              // type = 2 ("image")

    i32x8 g1;
    g1[0] = (1 << 16);                    // data_size=2B; wg_mask=0; no pad/iter
    g1[1] = (F_ << 16);                   // tensor_dim0 = 32      (bits 79:48 lo)
    g1[2] = (int)((unsigned)N_ << 16);    // tensor_dim1[15:0]=50000 (bits 111:80 lo)
    g1[3] = (F_ << 16);                   // tile_dim0 = 32        (bits 127:112)
    g1[4] = S_;                           // tile_dim1 = 16 valid indices
    g1[5] = F_;                           // tensor_dim0_stride = 32 (bits 191:160)
    g1[6] = 0;
    g1[7] = 0;

    i32x4 g2, g3;                         // 16 packed 16-bit row indices
    g2[0] = __builtin_amdgcn_readfirstlane((qa.x & 0xFFFF) | (qa.y << 16));
    g2[1] = __builtin_amdgcn_readfirstlane((qa.z & 0xFFFF) | (qa.w << 16));
    g2[2] = __builtin_amdgcn_readfirstlane((qb.x & 0xFFFF) | (qb.y << 16));
    g2[3] = __builtin_amdgcn_readfirstlane((qb.z & 0xFFFF) | (qb.w << 16));
    g3[0] = __builtin_amdgcn_readfirstlane((qc.x & 0xFFFF) | (qc.y << 16));
    g3[1] = __builtin_amdgcn_readfirstlane((qc.z & 0xFFFF) | (qc.w << 16));
    g3[2] = __builtin_amdgcn_readfirstlane((qd.x & 0xFFFF) | (qd.y << 16));
    g3[3] = __builtin_amdgcn_readfirstlane((qd.z & 0xFFFF) | (qd.w << 16));

    i32x8 gx = {0, 0, 0, 0, 0, 0, 0, 0};  // 6-arg toolchain: extension block, unused

    __builtin_amdgcn_tensor_load_to_lds(g0, g1, g2, g3, gx, /*cpol=*/0);
}

// --- main kernel: one WG (4 waves) per 16-row tile ---------------------------
template <bool XBF>
__global__ __launch_bounds__(128)
void spiralconv_wmma_kernel(const float* __restrict__ x,
                            const unsigned short* __restrict__ xbf,
                            const int* __restrict__ adj,
                            const unsigned short* __restrict__ Wbf,
                            const float* __restrict__ bias,
                            float* __restrict__ out) {
    __shared__ __align__(16) unsigned short ldsA[16 * LDS_STRIDE];

    const int wg    = blockIdx.x;
    const int batch = wg / TILES_PER_BATCH;
    const int tile  = wg - batch * TILES_PER_BATCH;
    const int row0  = tile * 16;
    const int tid   = threadIdx.x;
    const int wave  = tid >> 5;           // 0..3
    const int lane  = tid & 31;

    // ---------------- gather phase ------------------------------------------
    if (XBF) {
        // TDM gather mode: each wave issues 4 descriptors (one per tile row).
        const unsigned lds_base =
            (unsigned)(uintptr_t)(void*)&ldsA[0];
        const unsigned long long gaddr =
            (unsigned long long)(uintptr_t)(xbf + (size_t)batch * N_ * F_);
#pragma unroll
        for (int rr = 0; rr < 4; ++rr) {
            const int r = wave * 4 + rr;
            const int* arow = adj + ((size_t)batch * N_ + (size_t)(row0 + r)) * S_;
            tdm_gather_row(lds_base + (unsigned)(r * LDS_STRIDE * 2), gaddr, arow);
        }
        __builtin_amdgcn_s_wait_tensorcnt(0);
    } else {
        // fallback: VGPR gather from f32 x with inline bf16 conversion
#pragma unroll
        for (int i = 0; i < 2; ++i) {
            const int p = tid + i * 128;
            const int r = p >> 4;
            const int s = p & 15;
            const int nb = adj[((size_t)batch * N_ + (size_t)(row0 + r)) * S_ + s];
            unsigned short* dst = &ldsA[r * LDS_STRIDE + s * F_];
            const float4* src =
                (const float4*)(x + ((size_t)batch * N_ + (size_t)nb) * F_);
#pragma unroll
            for (int j = 0; j < 8; ++j) {
                float4 v = src[j];
                ushort4 o;
                o.x = f2bf(v.x); o.y = f2bf(v.y); o.z = f2bf(v.z); o.w = f2bf(v.w);
                ((ushort4*)dst)[j] = o;
            }
        }
    }
    __syncthreads();

    // ---------------- WMMA phase: wave w -> output columns [16w, 16w+16) -----
    const int half = lane >> 4;           // 0: lanes 0-15, 1: lanes 16-31
    const int l    = lane & 15;
    const int col  = wave * 16 + l;

    // A (16x32 bf16): lane l<16 = row l, K {kb..kb+15};
    //                 lanes 16-31 = same rows, K offsets +16.
    const unsigned short* Arow = &ldsA[l * LDS_STRIDE];
    // B (32x16 bf16): lane n holds column n, 16 contiguous K values
    //                 = 32 contiguous bytes of row-major W[col].
    const unsigned short* Wrow = Wbf + (size_t)col * K_;

    v8f acc = {0.f, 0.f, 0.f, 0.f, 0.f, 0.f, 0.f, 0.f};
#pragma unroll
    for (int k = 0; k < 16; ++k) {
        const int kb = k * 32;
        v8bf a0 = *(const v8bf*)(const void*)(Arow + kb + half * 16);
        v8bf a1 = *(const v8bf*)(const void*)(Arow + kb + half * 16 + 8);
        v16bf a = __builtin_shufflevector(a0, a1,
                    0,1,2,3,4,5,6,7,8,9,10,11,12,13,14,15);
        v16bf bm = *(const v16bf*)(const void*)(Wrow + kb + half * 16);
        acc = __builtin_amdgcn_wmma_f32_16x16x32_bf16(
                  false, a, false, bm, (short)0, acc, false, false);
    }

    // ---------------- epilogue: bias + ELU + last-row mask + store ------------
    const float bv = bias[col];
#pragma unroll
    for (int r = 0; r < 8; ++r) {
        const int rowN = row0 + r + half * 8;   // C/D: VGPR r -> M=r (+8 hi half)
        float v = acc[r] + bv;
        v = v > 0.f ? v : expm1f(v);
        if (rowN == N_ - 1) v = 0.f;            // zero padding vertex
        out[((size_t)batch * N_ + (size_t)rowN) * OUT_ + col] = v;
    }
}

// ---------------------------------------------------------------------------
extern "C" void kernel_launch(void* const* d_in, const int* in_sizes, int n_in,
                              void* d_out, int out_size, void* d_ws, size_t ws_size,
                              hipStream_t stream) {
    const float* x    = (const float*)d_in[0];
    const int*   adj  = (const int*)d_in[1];
    const float* W    = (const float*)d_in[2];
    const float* bias = (const float*)d_in[3];
    float*       out  = (float*)d_out;

    unsigned short* Wbf = (unsigned short*)d_ws;                       // 64 KB
    unsigned short* xbf = (unsigned short*)((char*)d_ws + 64 * 1024);  // 25.6 MB
    const size_t x_elems = (size_t)B_ * N_ * F_;
    const size_t need    = 64 * 1024 + x_elems * sizeof(unsigned short);

    convert_w_kernel<<<(OUT_ * K_) / 256, 256, 0, stream>>>(W, Wbf);

    const int nblocks = B_ * TILES_PER_BATCH;   // 25000
    if (ws_size >= need) {
        const int n4 = (int)(x_elems / 4);
        convert_x_kernel<<<(n4 + 255) / 256, 256, 0, stream>>>(x, xbf, n4);
        spiralconv_wmma_kernel<true><<<nblocks, 128, 0, stream>>>(
            x, xbf, adj, Wbf, bias, out);
    } else {
        spiralconv_wmma_kernel<false><<<nblocks, 128, 0, stream>>>(
            x, (const unsigned short*)nullptr, adj, Wbf, bias, out);
    }
}
